// ZetaOrganismLSTM_71433896067267
// MI455X (gfx1250) — matrix operand order repaired and checked by reading, not compiled, not run on hardware
//
#include <hip/hip_runtime.h>
#include <math.h>

// ---------------- types ----------------
typedef _Float16 v8h  __attribute__((ext_vector_type(8)));
typedef _Float16 v16h __attribute__((ext_vector_type(16)));
typedef float    v8f  __attribute__((ext_vector_type(8)));

#define N_PTS   8192
#define D_DIM   64
#define H_DIM   128
#define GRID_W  256
#define NCELLS  (GRID_W * GRID_W)
#define KTOT    256          // K of the gates GEMM: [states | nbr_mean | h]
#define G4H     512          // 4*H

// ---------------- workspace layout (bytes) ----------------
constexpr size_t OFF_COUNT  = 0;
constexpr size_t OFF_START  = OFF_COUNT  + (size_t)NCELLS * 4;
constexpr size_t OFF_CURSOR = OFF_START  + (size_t)NCELLS * 4;
constexpr size_t OFF_IDS    = OFF_CURSOR + (size_t)NCELLS * 4;
constexpr size_t OFF_XCAT   = OFF_IDS    + (size_t)N_PTS  * 4;           // f16 [N,256]
constexpr size_t OFF_WCAT   = OFF_XCAT   + (size_t)N_PTS * KTOT * 2;     // f16 [512,256]
constexpr size_t OFF_WOT    = OFF_WCAT   + (size_t)G4H * KTOT * 2;       // f16 [64,128] = W_out^T
constexpr size_t OFF_HNEW   = OFF_WOT    + (size_t)D_DIM * H_DIM * 2;    // f16 [N,128]
constexpr size_t WS_NEEDED  = OFF_HNEW   + (size_t)N_PTS * H_DIM * 2;

__device__ __forceinline__ float sigf(float x) { return 1.0f / (1.0f + __expf(-x)); }

// ---------------- binning ----------------
__global__ void k_zero(int* __restrict__ p, int n) {
    int i = blockIdx.x * blockDim.x + threadIdx.x;
    if (i < n) p[i] = 0;
}

__global__ void k_count(const int* __restrict__ pos, int* __restrict__ cnt) {
    int i = blockIdx.x * blockDim.x + threadIdx.x;
    if (i < N_PTS) atomicAdd(&cnt[pos[2 * i + 1] * GRID_W + pos[2 * i]], 1);
}

__global__ __launch_bounds__(1024) void k_scan(const int* __restrict__ cnt,
                                               int* __restrict__ start,
                                               int* __restrict__ cursor) {
    __shared__ int sh[1024];
    const int t = threadIdx.x;
    const int base = t * 64;
    int s = 0;
    for (int k = 0; k < 64; ++k) s += cnt[base + k];
    sh[t] = s;
    __syncthreads();
    for (int off = 1; off < 1024; off <<= 1) {
        int v = (t >= off) ? sh[t - off] : 0;
        __syncthreads();
        sh[t] += v;
        __syncthreads();
    }
    int run = (t == 0) ? 0 : sh[t - 1];
    for (int k = 0; k < 64; ++k) {
        int cv = cnt[base + k];
        start[base + k]  = run;
        cursor[base + k] = run;
        run += cv;
    }
}

__global__ void k_scatter(const int* __restrict__ pos, int* __restrict__ cursor,
                          int* __restrict__ ids) {
    int i = blockIdx.x * blockDim.x + threadIdx.x;
    if (i >= N_PTS) return;
    int cell = pos[2 * i + 1] * GRID_W + pos[2 * i];
    int slot = atomicAdd(&cursor[cell], 1);
    ids[slot] = i;
}

// insertion-sort each cell's id list -> deterministic accumulation order
__global__ void k_sortcells(const int* __restrict__ start, const int* __restrict__ cnt,
                            int* __restrict__ ids) {
    int cell = blockIdx.x * blockDim.x + threadIdx.x;
    if (cell >= NCELLS) return;
    int s = start[cell], n = cnt[cell];
    for (int a = 1; a < n; ++a) {
        int key = ids[s + a];
        int b = a - 1;
        while (b >= 0 && ids[s + b] > key) { ids[s + b + 1] = ids[s + b]; --b; }
        ids[s + b + 1] = key;
    }
}

// ---------------- neighbor mean + Xcat packing ----------------
// one block (128 threads) per point; threads 0..63 own one state dim each
__global__ __launch_bounds__(128) void k_neighbor_pack(
    const int* __restrict__ pos, const float* __restrict__ states,
    const float* __restrict__ h, const int* __restrict__ cstart,
    const int* __restrict__ ccount, const int* __restrict__ ids,
    _Float16* __restrict__ Xcat) {
    const int i = blockIdx.x;
    const int t = threadIdx.x;
    const int px = pos[2 * i], py = pos[2 * i + 1];
    _Float16* Xr = Xcat + (size_t)i * KTOT;
    if (t < D_DIM) {
        float acc = 0.0f;
        int cntn = 0;
        for (int dy = -3; dy <= 3; ++dy) {
            int cy = py + dy;
            if ((unsigned)cy >= GRID_W) continue;
            for (int dx = -3; dx <= 3; ++dx) {
                if (dx * dx + dy * dy > 9) continue;  // radius^2 = 9
                int cx = px + dx;
                if ((unsigned)cx >= GRID_W) continue;
                int cell = cy * GRID_W + cx;
                int s = cstart[cell], n = ccount[cell];
                for (int k = 0; k < n; ++k) {
                    int j = ids[s + k];
                    if (j == i) continue;  // exclude self (eye mask)
                    acc += states[(size_t)j * D_DIM + t];
                    ++cntn;
                }
            }
        }
        Xr[t]         = (_Float16)states[(size_t)i * D_DIM + t];
        Xr[D_DIM + t] = (_Float16)(acc / fmaxf((float)cntn, 1.0f));
    }
    Xr[2 * D_DIM + t] = (_Float16)h[(size_t)i * H_DIM + t];
}

// ---------------- weight packing ----------------
__global__ void k_pack_wcat(const float* __restrict__ Wih, const float* __restrict__ Whh,
                            _Float16* __restrict__ Wcat) {
    int idx = blockIdx.x * blockDim.x + threadIdx.x;   // 512*256
    if (idx >= G4H * KTOT) return;
    int r = idx >> 8, j = idx & 255;
    float v = (j < H_DIM) ? Wih[r * H_DIM + j] : Whh[r * H_DIM + (j - H_DIM)];
    Wcat[idx] = (_Float16)v;
}

__global__ void k_pack_woT(const float* __restrict__ Wout, _Float16* __restrict__ WoT) {
    int idx = blockIdx.x * blockDim.x + threadIdx.x;   // 64*128
    if (idx >= D_DIM * H_DIM) return;
    int n = idx >> 7, k = idx & 127;
    WoT[idx] = (_Float16)Wout[k * D_DIM + n];          // W_out^T [64,128]
}

// ---------------- fused gates GEMM + LSTM cell (WMMA f16->f32) ----------------
// Each wave computes a 16(M) x 16(H-cols) tile of i/f/g/o in 4 f32 accumulators.
// A fragment (16x32 f16): lane l%16 = row; per-lane contiguous K-run at 16*(l>>4).
// B fragment (32x16 f16): lane l%16 = col; per-lane contiguous K-run at 16*(l>>4).
// C/D (16x16 f32): lane l -> col = l%16; VGPR r -> row = r + 8*(l>>4).
__global__ __launch_bounds__(256) void k_gates_lstm(
    const _Float16* __restrict__ Xcat, const _Float16* __restrict__ Wcat,
    const float* __restrict__ bias, const float* __restrict__ c,
    _Float16* __restrict__ hnew) {
    const int lane = threadIdx.x & 31;
    const int wav  = threadIdx.x >> 5;
    const int tile = blockIdx.x * 8 + wav;     // 4096 tiles = 512 m-tiles x 8 n-tiles
    const int tm = tile >> 3;
    const int tn = tile & 7;
    const int m0 = tm << 4, n0 = tn << 4;
    const int l15 = lane & 15;
    const int hi  = lane >> 4;

    const _Float16* A  = Xcat + (size_t)(m0 + l15) * KTOT + hi * 16;
    const _Float16* B0 = Wcat + (size_t)(0 * H_DIM + n0 + l15) * KTOT + hi * 16;
    const _Float16* B1 = Wcat + (size_t)(1 * H_DIM + n0 + l15) * KTOT + hi * 16;
    const _Float16* B2 = Wcat + (size_t)(2 * H_DIM + n0 + l15) * KTOT + hi * 16;
    const _Float16* B3 = Wcat + (size_t)(3 * H_DIM + n0 + l15) * KTOT + hi * 16;

    v8f acc0 = {}, acc1 = {}, acc2 = {}, acc3 = {};
#pragma unroll
    for (int k0 = 0; k0 < KTOT; k0 += 32) {
        v8h alo = *(const v8h*)(A + k0);
        v8h ahi = *(const v8h*)(A + k0 + 8);
        v16h Av = __builtin_shufflevector(alo, ahi, 0,1,2,3,4,5,6,7,8,9,10,11,12,13,14,15);

        v8h bl, bh; v16h Bv;
        bl = *(const v8h*)(B0 + k0); bh = *(const v8h*)(B0 + k0 + 8);
        Bv = __builtin_shufflevector(bl, bh, 0,1,2,3,4,5,6,7,8,9,10,11,12,13,14,15);
        acc0 = __builtin_amdgcn_wmma_f32_16x16x32_f16(false, Av, false, Bv, (short)0, acc0, false, false);
        bl = *(const v8h*)(B1 + k0); bh = *(const v8h*)(B1 + k0 + 8);
        Bv = __builtin_shufflevector(bl, bh, 0,1,2,3,4,5,6,7,8,9,10,11,12,13,14,15);
        acc1 = __builtin_amdgcn_wmma_f32_16x16x32_f16(false, Av, false, Bv, (short)0, acc1, false, false);
        bl = *(const v8h*)(B2 + k0); bh = *(const v8h*)(B2 + k0 + 8);
        Bv = __builtin_shufflevector(bl, bh, 0,1,2,3,4,5,6,7,8,9,10,11,12,13,14,15);
        acc2 = __builtin_amdgcn_wmma_f32_16x16x32_f16(false, Av, false, Bv, (short)0, acc2, false, false);
        bl = *(const v8h*)(B3 + k0); bh = *(const v8h*)(B3 + k0 + 8);
        Bv = __builtin_shufflevector(bl, bh, 0,1,2,3,4,5,6,7,8,9,10,11,12,13,14,15);
        acc3 = __builtin_amdgcn_wmma_f32_16x16x32_f16(false, Av, false, Bv, (short)0, acc3, false, false);
    }

    const float bi = bias[0 * H_DIM + n0 + l15];
    const float bf = bias[1 * H_DIM + n0 + l15];
    const float bg = bias[2 * H_DIM + n0 + l15];
    const float bo = bias[3 * H_DIM + n0 + l15];
    const int mb = m0 + hi * 8;
    const int nc = n0 + l15;
#pragma unroll
    for (int r = 0; r < 8; ++r) {
        float cv = c[(size_t)(mb + r) * H_DIM + nc];
        float ig = sigf(acc0[r] + bi);
        float fg = sigf(acc1[r] + bf);
        float gg = tanhf(acc2[r] + bg);
        float og = sigf(acc3[r] + bo);
        float cn = fg * cv + ig * gg;
        float hn = og * tanhf(cn);
        hnew[(size_t)(mb + r) * H_DIM + nc] = (_Float16)hn;
    }
}

// ---------------- output head GEMM: new_states = h_new @ W_out (WMMA) ----------------
__global__ __launch_bounds__(256) void k_out_gemm(
    const _Float16* __restrict__ hnew, const _Float16* __restrict__ WoT,
    float* __restrict__ out) {
    const int lane = threadIdx.x & 31;
    const int wav  = threadIdx.x >> 5;
    const int tile = blockIdx.x * 8 + wav;     // 2048 tiles = 512 m x 4 n
    const int tm = tile >> 2;
    const int tn = tile & 3;
    const int m0 = tm << 4, n0 = tn << 4;
    const int l15 = lane & 15;
    const int hi  = lane >> 4;

    const _Float16* A = hnew + (size_t)(m0 + l15) * H_DIM + hi * 16;
    const _Float16* B = WoT  + (size_t)(n0 + l15) * H_DIM + hi * 16;

    v8f acc = {};
#pragma unroll
    for (int k0 = 0; k0 < H_DIM; k0 += 32) {
        v8h alo = *(const v8h*)(A + k0);
        v8h ahi = *(const v8h*)(A + k0 + 8);
        v16h Av = __builtin_shufflevector(alo, ahi, 0,1,2,3,4,5,6,7,8,9,10,11,12,13,14,15);
        v8h bl = *(const v8h*)(B + k0);
        v8h bh = *(const v8h*)(B + k0 + 8);
        v16h Bv = __builtin_shufflevector(bl, bh, 0,1,2,3,4,5,6,7,8,9,10,11,12,13,14,15);
        acc = __builtin_amdgcn_wmma_f32_16x16x32_f16(false, Av, false, Bv, (short)0, acc, false, false);
    }
    const int mb = m0 + hi * 8;
    const int nc = n0 + l15;
#pragma unroll
    for (int r = 0; r < 8; ++r)
        out[(size_t)(mb + r) * D_DIM + nc] = acc[r];
}

// ---------------- role head + softmax (3-wide, trivial VALU) ----------------
__global__ void k_role(const _Float16* __restrict__ hnew, const float* __restrict__ Wr,
                       float* __restrict__ out) {
    int i = blockIdx.x * blockDim.x + threadIdx.x;
    if (i >= N_PTS) return;
    const _Float16* hr = hnew + (size_t)i * H_DIM;
    float a0 = 0.f, a1 = 0.f, a2 = 0.f;
#pragma unroll 4
    for (int k = 0; k < H_DIM; ++k) {
        float hv = (float)hr[k];
        a0 += hv * Wr[k * 3 + 0];
        a1 += hv * Wr[k * 3 + 1];
        a2 += hv * Wr[k * 3 + 2];
    }
    float m = fmaxf(a0, fmaxf(a1, a2));
    float e0 = __expf(a0 - m), e1 = __expf(a1 - m), e2 = __expf(a2 - m);
    float inv = 1.0f / (e0 + e1 + e2);
    out[i * 3 + 0] = e0 * inv;
    out[i * 3 + 1] = e1 * inv;
    out[i * 3 + 2] = e2 * inv;
}

// ---------------- launch ----------------
extern "C" void kernel_launch(void* const* d_in, const int* in_sizes, int n_in,
                              void* d_out, int out_size, void* d_ws, size_t ws_size,
                              hipStream_t stream) {
    (void)in_sizes; (void)n_in; (void)out_size;
    if (ws_size < WS_NEEDED) return;

    const int*   positions = (const int*)d_in[0];
    const float* states    = (const float*)d_in[1];
    const float* h         = (const float*)d_in[2];
    const float* c         = (const float*)d_in[3];
    const float* Wih       = (const float*)d_in[4];
    const float* Whh       = (const float*)d_in[5];
    const float* b         = (const float*)d_in[6];
    const float* Wout      = (const float*)d_in[7];
    const float* Wrole     = (const float*)d_in[8];

    char* ws = (char*)d_ws;
    int*       cell_count  = (int*)(ws + OFF_COUNT);
    int*       cell_start  = (int*)(ws + OFF_START);
    int*       cell_cursor = (int*)(ws + OFF_CURSOR);
    int*       ids         = (int*)(ws + OFF_IDS);
    _Float16*  Xcat        = (_Float16*)(ws + OFF_XCAT);
    _Float16*  Wcat        = (_Float16*)(ws + OFF_WCAT);
    _Float16*  WoT         = (_Float16*)(ws + OFF_WOT);
    _Float16*  hnew        = (_Float16*)(ws + OFF_HNEW);

    float* out_states = (float*)d_out;                             // [N,64]
    float* out_roles  = (float*)d_out + (size_t)N_PTS * D_DIM;     // [N,3]

    // spatial binning (deterministic via per-cell sort)
    k_zero<<<NCELLS / 256, 256, 0, stream>>>(cell_count, NCELLS);
    k_count<<<N_PTS / 256, 256, 0, stream>>>(positions, cell_count);
    k_scan<<<1, 1024, 0, stream>>>(cell_count, cell_start, cell_cursor);
    k_scatter<<<N_PTS / 256, 256, 0, stream>>>(positions, cell_cursor, ids);
    k_sortcells<<<NCELLS / 256, 256, 0, stream>>>(cell_start, cell_count, ids);

    // neighbor mean + activation/weight packing (f32 -> f16)
    k_neighbor_pack<<<N_PTS, 128, 0, stream>>>(positions, states, h, cell_start,
                                               cell_count, ids, Xcat);
    k_pack_wcat<<<(G4H * KTOT) / 256, 256, 0, stream>>>(Wih, Whh, Wcat);
    k_pack_woT<<<(D_DIM * H_DIM) / 256, 256, 0, stream>>>(Wout, WoT);

    // fused 4-gate GEMM + LSTM cell, then output heads
    k_gates_lstm<<<512, 256, 0, stream>>>(Xcat, Wcat, b, c, hnew);
    k_out_gemm<<<256, 256, 0, stream>>>(hnew, WoT, out_states);
    k_role<<<N_PTS / 256, 256, 0, stream>>>(hnew, Wrole, out_roles);
}
